// IsingTransformerLayer_7876970021364
// MI455X (gfx1250) — compile-verified, matrix operand order checked by reading, not run on hardware
//
#include <hip/hip_runtime.h>

// ---------------------------------------------------------------------------
// Problem constants (from reference)
// ---------------------------------------------------------------------------
static const int NT   = 1024;   // sequence length N
static const int DIMD = 1024;   // model dim
static const int DH   = 64;     // head dim
static const int MH   = 5;      // Anderson history
static const int FITERS = 40;   // Anderson iterations
static const int SOLVE_STEPS = 3;  // Neumann/Jacobi sweeps inside the loop
static const int FINAL_STEPS = 4;  // sweeps for the final gradient solve
#define EPSV   1e-5f
#define RIDGEV 1e-5f
#define INV_SQRT_DIM (1.0f/32.0f)

typedef __bf16 bf16;
typedef __attribute__((ext_vector_type(16))) __bf16 v16bf;
typedef __attribute__((ext_vector_type(8)))  __bf16 v8bf;
typedef __attribute__((ext_vector_type(8)))  float  v8f;
typedef __attribute__((ext_vector_type(4)))  int    v4i;

typedef __attribute__((address_space(1))) v4i as1_v4i;
typedef __attribute__((address_space(3))) v4i as3_v4i;

// --------------------------- async copy to LDS -----------------------------
#if defined(__has_builtin)
# if __has_builtin(__builtin_amdgcn_global_load_async_to_lds_b128)
#  define USE_ASYNC_LDS 1
# endif
#endif
#ifndef USE_ASYNC_LDS
# define USE_ASYNC_LDS 0
#endif

__device__ __forceinline__ void cp_b128_to_lds(const bf16* g, bf16* s) {
#if USE_ASYNC_LDS
  __builtin_amdgcn_global_load_async_to_lds_b128(
      (as1_v4i*)(uintptr_t)g,   // global source (AS1, v4i = 16B)
      (as3_v4i*)s,              // LDS destination (AS3 via addrspacecast)
      0, 0);
#else
  *(v8bf*)s = *(const v8bf*)g;   // sync fallback: global_load_b128 + ds_store
#endif
}

template <int P>
__device__ __forceinline__ void wait_async() {
#if USE_ASYNC_LDS
# if defined(__has_builtin) && __has_builtin(__builtin_amdgcn_s_wait_asynccnt)
  __builtin_amdgcn_s_wait_asynccnt(P);
# else
  asm volatile("s_wait_asynccnt %0" :: "n"(P) : "memory");
# endif
#endif
}

__device__ __forceinline__ v8f wmma_bf16(v16bf a, v16bf b, v8f c) {
  return __builtin_amdgcn_wmma_f32_16x16x32_bf16(false, a, false, b,
                                                 (short)0, c, false, false);
}

// ---------------------------------------------------------------------------
// LayerNorm: h = ((x-mu)/sqrt(var+eps)*w + b) / sqrt(DIM)
// ---------------------------------------------------------------------------
__global__ void k_layernorm(const float* __restrict__ x, const float* __restrict__ w,
                            const float* __restrict__ b, float* __restrict__ h) {
  __shared__ float red[256];
  const int row = blockIdx.x, tid = threadIdx.x;
  const float* xr = x + row * DIMD;
  float s = 0.f;
  for (int j = tid; j < DIMD; j += 256) s += xr[j];
  red[tid] = s; __syncthreads();
  for (int o = 128; o > 0; o >>= 1) { if (tid < o) red[tid] += red[tid + o]; __syncthreads(); }
  const float mu = red[0] / (float)DIMD;
  __syncthreads();
  float v = 0.f;
  for (int j = tid; j < DIMD; j += 256) { float d = xr[j] - mu; v += d * d; }
  red[tid] = v; __syncthreads();
  for (int o = 128; o > 0; o >>= 1) { if (tid < o) red[tid] += red[tid + o]; __syncthreads(); }
  const float rstd = rsqrtf(red[0] / (float)DIMD + EPSV);
  for (int j = tid; j < DIMD; j += 256)
    h[row * DIMD + j] = ((xr[j] - mu) * rstd * w[j] + b[j]) * INV_SQRT_DIM;
}

// ---------------------------------------------------------------------------
// q/k projection + RoPE (q carries DHEAD^-0.5 * sqrt(DIM) = 4)
// ---------------------------------------------------------------------------
__global__ void k_qk_rope(const float* __restrict__ h, const float* __restrict__ Wq,
                          const float* __restrict__ Wk, const float* __restrict__ sinp,
                          const float* __restrict__ cosp,
                          float* __restrict__ q, float* __restrict__ k) {
  __shared__ float hr[DIMD];
  __shared__ float qr[DH], kr[DH];
  const int row = blockIdx.x, tid = threadIdx.x;  // 64 threads
  for (int j = tid; j < DIMD; j += 64) hr[j] = h[row * DIMD + j];
  __syncthreads();
  const float* wq = Wq + tid * DIMD;
  const float* wk = Wk + tid * DIMD;
  float sq = 0.f, sk = 0.f;
  for (int j = 0; j < DIMD; ++j) { const float hv = hr[j]; sq += hv * wq[j]; sk += hv * wk[j]; }
  qr[tid] = sq; kr[tid] = sk;
  __syncthreads();
  const float cs = cosp[row * DH + tid];
  const float sn = sinp[row * DH + tid];
  const float rotq = (tid & 1) ? qr[tid - 1] : -qr[tid + 1];
  const float rotk = (tid & 1) ? kr[tid - 1] : -kr[tid + 1];
  q[row * DH + tid] = (qr[tid] * cs + rotq * sn) * 4.0f;
  k[row * DH + tid] =  kr[tid] * cs + rotk * sn;
}

// ---------------------------------------------------------------------------
// Causal softmax row -> J = attn/1024 (f32 + bf16 + bf16-transposed)
// ---------------------------------------------------------------------------
__global__ void k_attn(const float* __restrict__ q, const float* __restrict__ k,
                       float* __restrict__ Jf, bf16* __restrict__ Jbf,
                       bf16* __restrict__ Jtbf) {
  __shared__ float sv[NT];
  __shared__ float qs[DH];
  __shared__ float red[256];
  const int i = blockIdx.x, tid = threadIdx.x;
  for (int d = tid; d < DH; d += 256) qs[d] = q[i * DH + d];
  __syncthreads();
  float mx = -1e30f;
  for (int j = tid; j < NT; j += 256) {
    float s = -1e30f;
    if (j <= i) {
      s = 0.f;
      const float* kj = k + j * DH;
      for (int d = 0; d < DH; ++d) s += qs[d] * kj[d];
    }
    sv[j] = s;
    if (s > mx) mx = s;
  }
  red[tid] = mx; __syncthreads();
  for (int o = 128; o > 0; o >>= 1) { if (tid < o) red[tid] = fmaxf(red[tid], red[tid + o]); __syncthreads(); }
  mx = red[0]; __syncthreads();
  float sum = 0.f;
  for (int j = tid; j < NT; j += 256) {
    const float e = (j <= i) ? __expf(sv[j] - mx) : 0.f;
    sv[j] = e; sum += e;
  }
  red[tid] = sum; __syncthreads();
  for (int o = 128; o > 0; o >>= 1) { if (tid < o) red[tid] += red[tid + o]; __syncthreads(); }
  const float inv = 1.0f / (red[0] * 1024.0f);   // sqrt(DIM*N) = 1024
  for (int j = tid; j < NT; j += 256) {
    const float p = sv[j] * inv;
    Jf[i * NT + j]   = p;
    Jbf[i * NT + j]  = static_cast<bf16>(p);
    Jtbf[j * NT + i] = static_cast<bf16>(p);
  }
}

// ---------------------------------------------------------------------------
// Fused WMMA GEMM + Jacobi update:
//   yout = (h + A @ y) * tinv   ;  A (NT x NT bf16),  y given transposed as
//   yTin (DIMD x NT bf16).  Writes yToutT (bf16, transposed) and optionally
//   y (f32, row-major, for the gradient reductions).
//
// Workgroup: 128 thr (4 waves), C-tile 64(M) x 128(N); wave-tile 32 x 64
// (2 A-frags x 4 B-frags = 8 WMMAs / K-step).  A/B K-slices (64x32, 128x32)
// are double-buffered in LDS via GLOBAL_LOAD_ASYNC_TO_LDS_B128 (ASYNCcnt),
// row pitch padded to 80 B so fragment ds_load_b128s are bank-balanced.
// Epilogue: each C fragment holds 8 consecutive M per lane, so the
// transposed bf16 store is a single contiguous b128 per fragment.
// ---------------------------------------------------------------------------
#define LDA_P 40   // LDS row pitch in bf16 (32 data + 8 pad)

__global__ __launch_bounds__(128) void k_gemm_fused(
    const bf16* __restrict__ A, const bf16* __restrict__ BT,
    const float* __restrict__ h, const float* __restrict__ tinv,
    float* __restrict__ yF, bf16* __restrict__ yTout) {
  __shared__ __align__(16) bf16 sA[2][64 * LDA_P];
  __shared__ __align__(16) bf16 sB[2][128 * LDA_P];
  const int tid  = threadIdx.x;
  const int bm   = blockIdx.x >> 3;        // 0..15  (M blocks of 64)
  const int bn   = blockIdx.x & 7;         // 0..7   (N blocks of 128)
  const int wave = tid >> 5, lane = tid & 31;
  const int wm   = wave >> 1, wn = wave & 1;
  const int l    = lane & 15, half = lane >> 4;

  const bf16* gA = A  + (size_t)(bm * 64)  * NT;   // 64 rows of A
  const bf16* gB = BT + (size_t)(bn * 128) * NT;   // 128 cols of B (rows of BT)

  v8f acc[2][4] = {};

  auto stage = [&](int buf, int k0) {
#pragma unroll
    for (int i = 0; i < 2; ++i) {            // A: 64 rows x 4 chunks = 256
      const int id = tid * 2 + i;
      cp_b128_to_lds(gA + (size_t)(id >> 2) * NT + k0 + (id & 3) * 8,
                     &sA[buf][(id >> 2) * LDA_P + (id & 3) * 8]);
    }
#pragma unroll
    for (int i = 0; i < 4; ++i) {            // B: 128 rows x 4 chunks = 512
      const int id = tid * 4 + i;
      cp_b128_to_lds(gB + (size_t)(id >> 2) * NT + k0 + (id & 3) * 8,
                     &sB[buf][(id >> 2) * LDA_P + (id & 3) * 8]);
    }
  };

  auto frag = [&](const bf16* s, int row) -> v16bf {
    const v8bf x0 = *(const v8bf*)(s + row * LDA_P + half * 8);
    const v8bf x1 = *(const v8bf*)(s + row * LDA_P + 16 + half * 8);
    return __builtin_shufflevector(x0, x1, 0,1,2,3,4,5,6,7,8,9,10,11,12,13,14,15);
  };

  stage(0, 0);
  for (int it = 0; it < NT / 32; ++it) {
    const int buf = it & 1;
    if (it < NT / 32 - 1) {
      stage(buf ^ 1, (it + 1) * 32);   // prefetch next K-slice (async)
      wait_async<6>();                 // current slice's 6 ops done (in-order)
    } else {
      wait_async<0>();
    }
    __syncthreads();
    v16bf af[2], bfv[4];
#pragma unroll
    for (int sa = 0; sa < 2; ++sa) af[sa]  = frag(sA[buf], wm * 32 + sa * 16 + l);
#pragma unroll
    for (int sb = 0; sb < 4; ++sb) bfv[sb] = frag(sB[buf], wn * 64 + sb * 16 + l);
#pragma unroll
    for (int sa = 0; sa < 2; ++sa)
#pragma unroll
      for (int sb = 0; sb < 4; ++sb)
        acc[sa][sb] = wmma_bf16(af[sa], bfv[sb], acc[sa][sb]);
    __syncthreads();   // all waves done reading buf before it is re-staged
  }

  // fused epilogue: y' = (h + C)*tinv ; transposed bf16 store is contiguous
  const int rbase = bm * 64 + wm * 32;
  const int cbase = bn * 128 + wn * 64;
#pragma unroll
  for (int sa = 0; sa < 2; ++sa) {
    const int m8 = rbase + sa * 16 + half * 8;   // 8 consecutive M per lane
#pragma unroll
    for (int sb = 0; sb < 4; ++sb) {
      const int c = cbase + sb * 16 + l;
      v8bf pk;
#pragma unroll
      for (int r = 0; r < 8; ++r) {
        const float v = (h[(size_t)(m8 + r) * DIMD + c] + acc[sa][sb][r]) * tinv[m8 + r];
        if (yF) yF[(size_t)(m8 + r) * DIMD + c] = v;
        pk[r] = static_cast<bf16>(v);
      }
      *(v8bf*)(yTout + (size_t)c * NT + m8) = pk;
    }
  }
}

// ---------------------------------------------------------------------------
// y0T = transpose(h * tinv) in bf16
// ---------------------------------------------------------------------------
__global__ void k_y0(const float* __restrict__ h, const float* __restrict__ tinv,
                     bf16* __restrict__ yT) {
  const int idx = blockIdx.x * 256 + threadIdx.x;
  const int n = idx >> 10, d = idx & 1023;
  yT[(size_t)d * NT + n] = static_cast<bf16>(h[idx] * tinv[n]);
}

// ---------------------------------------------------------------------------
// g_i = 1 - 0.5*diag(V^-1)_i - 0.25*sum_d yv_i,d*yw_i,d + t_i
// diag(V^-1)_i ~ (1 + J_ii/t_i + (sum_k J_ik J_ki / t_k)/t_i)/t_i
// ---------------------------------------------------------------------------
__global__ void k_gradt(const float* __restrict__ Jf, const bf16* __restrict__ Jtbf,
                        const float* __restrict__ t, const float* __restrict__ tinv,
                        const float* __restrict__ yv, const float* __restrict__ yw,
                        float* __restrict__ G, float* __restrict__ R,
                        int slot, int fillall) {
  __shared__ float r1[256], r2[256];
  const int i = blockIdx.x, tid = threadIdx.x;
  float s1 = 0.f, s2 = 0.f;
  for (int kx = tid; kx < NT; kx += 256)
    s1 += Jf[(size_t)i * NT + kx] * (float)Jtbf[(size_t)i * NT + kx] * tinv[kx];
  for (int d = tid; d < DIMD; d += 256)
    s2 += yv[(size_t)i * DIMD + d] * yw[(size_t)i * DIMD + d];
  r1[tid] = s1; r2[tid] = s2; __syncthreads();
  for (int o = 128; o > 0; o >>= 1) {
    if (tid < o) { r1[tid] += r1[tid + o]; r2[tid] += r2[tid + o]; }
    __syncthreads();
  }
  if (tid == 0) {
    const float ti = t[i], ri = tinv[i];
    const float Jii = Jf[(size_t)i * NT + i];
    const float diag = (1.f + Jii * ri + r1[0] * ri) * ri;
    const float g = 1.0f - 0.5f * diag - 0.25f * r2[0] + ti;   // BETA = 1
    if (fillall) {
      for (int m = 0; m < MH; ++m) { G[m * NT + i] = g; R[m * NT + i] = g - ti; }
    } else {
      G[slot * NT + i] = g; R[slot * NT + i] = g - ti;
    }
  }
}

// ---------------------------------------------------------------------------
// Anderson combine: gram = R R^T + ridge*I ; w = gram^-1 1 ; a = w/sum(w)
// ---------------------------------------------------------------------------
__global__ void k_gram_solve(const float* __restrict__ R, float* __restrict__ a) {
  __shared__ float red[256];
  __shared__ float gram[MH][MH];
  const int tid = threadIdx.x;
  for (int i = 0; i < MH; ++i)
    for (int j = i; j < MH; ++j) {
      float s = 0.f;
      for (int n = tid; n < NT; n += 256) s += R[i * NT + n] * R[j * NT + n];
      red[tid] = s; __syncthreads();
      for (int o = 128; o > 0; o >>= 1) { if (tid < o) red[tid] += red[tid + o]; __syncthreads(); }
      if (tid == 0) {
        float v = red[0]; if (i == j) v += RIDGEV;
        gram[i][j] = v; gram[j][i] = v;
      }
      __syncthreads();
    }
  if (tid == 0) {
    float Am[MH][MH], bv[MH], w[MH];
    for (int i = 0; i < MH; ++i) { bv[i] = 1.f; for (int j = 0; j < MH; ++j) Am[i][j] = gram[i][j]; }
    for (int c = 0; c < MH; ++c) {
      int p = c;
      for (int r = c + 1; r < MH; ++r) if (fabsf(Am[r][c]) > fabsf(Am[p][c])) p = r;
      if (p != c) {
        for (int cc = 0; cc < MH; ++cc) { float tv = Am[c][cc]; Am[c][cc] = Am[p][cc]; Am[p][cc] = tv; }
        float tb = bv[c]; bv[c] = bv[p]; bv[p] = tb;
      }
      const float piv = Am[c][c];
      for (int r = c + 1; r < MH; ++r) {
        const float f = Am[r][c] / piv;
        for (int cc = c; cc < MH; ++cc) Am[r][cc] -= f * Am[c][cc];
        bv[r] -= f * bv[c];
      }
    }
    for (int r = MH - 1; r >= 0; --r) {
      float s = bv[r];
      for (int cc = r + 1; cc < MH; ++cc) s -= Am[r][cc] * w[cc];
      w[r] = s / Am[r][r];
    }
    float sum = 0.f;
    for (int m = 0; m < MH; ++m) sum += w[m];
    for (int m = 0; m < MH; ++m) a[m] = w[m] / sum;
  }
}

__global__ void k_combine(const float* __restrict__ a, const float* __restrict__ G,
                          float* __restrict__ t, float* __restrict__ tinv) {
  const int n = blockIdx.x * 256 + threadIdx.x;
  float s = 0.f;
#pragma unroll
  for (int m = 0; m < MH; ++m) s += a[m] * G[m * NT + n];
  t[n] = s;
  tinv[n] = 1.0f / s;
}

__global__ void k_fill(float* __restrict__ t, float* __restrict__ tinv, float v) {
  const int n = blockIdx.x * 256 + threadIdx.x;
  t[n] = v;
  tinv[n] = 1.0f / v;
}

__global__ void k_final(const float* __restrict__ yv, const float* __restrict__ yw,
                        float* __restrict__ out) {
  const int idx = blockIdx.x * 256 + threadIdx.x;
  out[idx] = -0.25f * (yv[idx] + yw[idx]);
}

// ---------------------------------------------------------------------------
// Host-side orchestration
// ---------------------------------------------------------------------------
static void eval_solves(const bf16* Abf, const float* h, const float* tinv,
                        bf16* yTa, bf16* yTb, float* yfinal, int steps,
                        hipStream_t st) {
  const int EW = (NT * DIMD) / 256;
  k_y0<<<EW, 256, 0, st>>>(h, tinv, yTa);
  bf16* cur = yTa; bf16* nxt = yTb;
  for (int s = 0; s < steps; ++s) {
    float* yF = (s == steps - 1) ? yfinal : nullptr;
    k_gemm_fused<<<128, 128, 0, st>>>(Abf, cur, h, tinv, yF, nxt);
    bf16* tmp = cur; cur = nxt; nxt = tmp;
  }
}

static void eval_f(const bf16* Jbf, const bf16* Jtbf, const float* Jf,
                   const float* h, const float* t, const float* tinv,
                   float* yv, float* yw, bf16* yTa, bf16* yTb,
                   float* G, float* R, int slot, int fillall, int steps,
                   hipStream_t st) {
  eval_solves(Jbf,  h, tinv, yTa, yTb, yv, steps, st);   // V^-1  h
  eval_solves(Jtbf, h, tinv, yTa, yTb, yw, steps, st);   // V^-T  h
  if (G) k_gradt<<<NT, 256, 0, st>>>(Jf, Jtbf, t, tinv, yv, yw, G, R, slot, fillall);
}

extern "C" void kernel_launch(void* const* d_in, const int* in_sizes, int n_in,
                              void* d_out, int out_size, void* d_ws, size_t ws_size,
                              hipStream_t stream) {
  (void)in_sizes; (void)n_in; (void)out_size; (void)ws_size;
  const float* x    = (const float*)d_in[0];
  const float* sinp = (const float*)d_in[1];
  const float* cosp = (const float*)d_in[2];
  const float* lnw  = (const float*)d_in[3];
  const float* lnb  = (const float*)d_in[4];
  const float* Wq   = (const float*)d_in[5];
  const float* Wk   = (const float*)d_in[6];
  float* out = (float*)d_out;

  char* base = (char*)d_ws;
  size_t off = 0;
  auto alloc = [&](size_t bytes) -> void* {
    void* p = base + off;
    off += (bytes + 255) & ~(size_t)255;
    return p;
  };
  float* h    = (float*)alloc((size_t)NT * DIMD * 4);
  float* Jf   = (float*)alloc((size_t)NT * NT * 4);
  float* yv   = (float*)alloc((size_t)NT * DIMD * 4);
  float* yw   = (float*)alloc((size_t)NT * DIMD * 4);
  bf16* Jbf   = (bf16*)alloc((size_t)NT * NT * 2);
  bf16* Jtbf  = (bf16*)alloc((size_t)NT * NT * 2);
  bf16* yTa   = (bf16*)alloc((size_t)DIMD * NT * 2);
  bf16* yTb   = (bf16*)alloc((size_t)DIMD * NT * 2);
  float* q    = (float*)alloc((size_t)NT * DH * 4);
  float* kk   = (float*)alloc((size_t)NT * DH * 4);
  float* G    = (float*)alloc((size_t)MH * NT * 4);
  float* R    = (float*)alloc((size_t)MH * NT * 4);
  float* t    = (float*)alloc((size_t)NT * 4);
  float* tinv = (float*)alloc((size_t)NT * 4);
  float* a    = (float*)alloc(64);

  // preamble: h, q/k + rope, attn -> J
  k_layernorm<<<NT, 256, 0, stream>>>(x, lnw, lnb, h);
  k_qk_rope<<<NT, 64, 0, stream>>>(h, Wq, Wk, sinp, cosp, q, kk);
  k_attn<<<NT, 256, 0, stream>>>(q, kk, Jf, Jbf, Jtbf);

  // Anderson acceleration of t = f(t), t0 = 0.5
  k_fill<<<NT / 256, 256, 0, stream>>>(t, tinv, 0.5f);
  eval_f(Jbf, Jtbf, Jf, h, t, tinv, yv, yw, yTa, yTb, G, R, 0, /*fillall=*/1,
         SOLVE_STEPS, stream);
  for (int it = 1; it < FITERS; ++it) {
    k_gram_solve<<<1, 256, 0, stream>>>(R, a);
    k_combine<<<NT / 256, 256, 0, stream>>>(a, G, t, tinv);
    eval_f(Jbf, Jtbf, Jf, h, t, tinv, yv, yw, yTa, yTb, G, R, it % MH, 0,
           SOLVE_STEPS, stream);
  }
  k_gram_solve<<<1, 256, 0, stream>>>(R, a);
  k_combine<<<NT / 256, 256, 0, stream>>>(a, G, t, tinv);   // t = t*

  // final gradient: out = -0.25 * (V^-1 h + V^-T h) at t*
  eval_f(Jbf, Jtbf, Jf, h, t, tinv, yv, yw, yTa, yTb, nullptr, nullptr, 0, 0,
         FINAL_STEPS, stream);
  k_final<<<(NT * DIMD) / 256, 256, 0, stream>>>(yv, yw, out);
}
